// LocalGraph_21646635172634
// MI455X (gfx1250) — matrix operand order repaired and checked by reading, not compiled, not run on hardware
//
#include <hip/hip_runtime.h>

// MI455X / gfx1250, wave32. Fused LocalGraph MLP stack, one block per (b,m) slice.
// B=8, M=128, N=256, D_IN=8, H=64.

typedef __attribute__((ext_vector_type(16))) __bf16 v16bf;
typedef __attribute__((ext_vector_type(8)))  float  v8f;

#define TPB 256
#define NWAVE 8

__device__ __forceinline__ unsigned short f2bf(float f) {
  unsigned u = __float_as_uint(f);
  unsigned r = u + 0x7FFFu + ((u >> 16) & 1u);   // round-to-nearest-even
  return (unsigned short)(r >> 16);
}
__device__ __forceinline__ float bf2f(unsigned short h) {
  return __uint_as_float(((unsigned)h) << 16);
}

union V16U {
  v16bf v;
  unsigned short u[16];
  uint4 q[2];
};

// Stage f32 weights (K x N, row-major) into LDS as bf16 in WMMA B-operand order:
// group (kb, n) holds K = kb*32 .. kb*32+31 contiguously; lane L reads
// [hi*8, hi*8+8) and [16+hi*8, 16+hi*8+8) of the group (interleaved 16-bit
// layout, same K order as A).
template<int K, int N>
__device__ __forceinline__ void stage_weights(const float* __restrict__ W,
                                              unsigned short* wlds) {
  for (int idx = threadIdx.x; idx < K * N; idx += TPB) {
    int k = idx / N, n = idx - k * N;
    wlds[(unsigned)((k >> 5) * N + n) * 32u + (unsigned)(k & 31)] = f2bf(W[idx]);
  }
}

// One Linear(+bias) -> LayerNorm -> ReLU stage on WMMA bf16.
// NT: number of 16-wide N tiles (Nout = NT*16). KB: number of 32-deep K blocks.
// AMODE 0: A straight from LDS activations (row-major, stride astride, bf16).
// AMODE 1: K blocks 0..1 from H2 directly; K blocks 2..3 are the on-the-fly
//          exclude-self-max concat half built from H2 + (m1,m2,i1).
// outH != nullptr: store LN+ReLU result bf16 row-major (stride Nout).
// outH == nullptr: reduce column-max into colmax[] (f32 bits as uint, vals >= 0).
template<int NT, int KB, int AMODE>
__device__ void mlp_stage(const unsigned short* __restrict__ aH, int astride,
                          const unsigned short* __restrict__ wlds,
                          const float* __restrict__ bias,
                          const float* __restrict__ gamma,
                          const float* __restrict__ beta,
                          unsigned short* __restrict__ outH,
                          const float* __restrict__ m1,
                          const float* __restrict__ m2,
                          const int* __restrict__ i1,
                          unsigned* __restrict__ colmax) {
  const int tid  = threadIdx.x;
  const int lane = tid & 31;
  const int wave = tid >> 5;
  const int l15  = lane & 15;
  const int hi   = lane >> 4;        // K-half / M-half selector
  const int Nout = NT * 16;

  for (int s = wave; s < 16; s += NWAVE) {   // 16 row-strips of 16, 8 waves
    const int m0 = s * 16;
    v8f acc[NT];
#pragma unroll
    for (int t = 0; t < NT; ++t) acc[t] = (v8f)0.0f;

    const int arow = m0 + l15;               // A: M = lane&15 in both halves

#pragma unroll
    for (int kb = 0; kb < KB; ++kb) {
      V16U A;
      if (AMODE == 1 && kb >= 2) {
        // concat half: neigh[i][f] = max((i==i1[f]?m2[f]:m1[f]), h2[i][f]-1e4)
        const int f0 = kb * 32 - 64;
#pragma unroll
        for (int e = 0; e < 16; ++e) {
          // 16-bit A layout: e<8 -> K=(hi?8:0)+e ; e>=8 -> K=16+(hi?8:0)+(e-8)
          int kk = ((e < 8) ? 0 : 16) + (hi ? 8 : 0) + (e & 7);
          int f  = f0 + kk;
          float h   = bf2f(aH[arow * astride + f]);
          float sel = (arow == i1[f]) ? m2[f] : m1[f];
          A.u[e] = f2bf(fmaxf(sel, h - 10000.0f));
        }
      } else {
        // two contiguous 16B loads matching interleaved 16-bit A layout
        const unsigned short* ap = aH + arow * astride + kb * 32 + hi * 8;
        A.q[0] = *(const uint4*)(ap);
        A.q[1] = *(const uint4*)(ap + 16);
      }
#pragma unroll
      for (int t = 0; t < NT; ++t) {
        V16U Bv;
        const unsigned short* bp =
            wlds + (unsigned)(kb * Nout + t * 16 + l15) * 32u + hi * 8;
        Bv.q[0] = *(const uint4*)(bp);
        Bv.q[1] = *(const uint4*)(bp + 16);
        acc[t] = __builtin_amdgcn_wmma_f32_16x16x32_bf16(
            false, A.v, false, Bv.v, (short)0, acc[t], false, false);
      }
    }

    // bias + LayerNorm(+ReLU). Row r lives in acc[*][r] across the 16 lanes of
    // this half-wave (lanes 0-15: rows m0+r, lanes 16-31: rows m0+8+r).
    float ps[8], ss[8];
#pragma unroll
    for (int r = 0; r < 8; ++r) { ps[r] = 0.0f; ss[r] = 0.0f; }
#pragma unroll
    for (int t = 0; t < NT; ++t) {
      float bv = bias[t * 16 + l15];
#pragma unroll
      for (int r = 0; r < 8; ++r) {
        float v = acc[t][r] + bv;
        acc[t][r] = v;
        ps[r] += v;
        ss[r] += v * v;
      }
    }
    for (int mask = 1; mask <= 8; mask <<= 1) {   // stays within 16-lane half
#pragma unroll
      for (int r = 0; r < 8; ++r) {
        ps[r] += __shfl_xor(ps[r], mask, 32);
        ss[r] += __shfl_xor(ss[r], mask, 32);
      }
    }
    const float invN = 1.0f / (float)Nout;
    float mu[8], rs[8];
#pragma unroll
    for (int r = 0; r < 8; ++r) {
      float m   = ps[r] * invN;
      float var = ss[r] * invN - m * m;
      mu[r] = m;
      rs[r] = rsqrtf(var + 1e-5f);
    }
#pragma unroll
    for (int t = 0; t < NT; ++t) {
      const int n = t * 16 + l15;
      const float g = gamma[n], be = beta[n];
      if (outH != nullptr) {
#pragma unroll
        for (int r = 0; r < 8; ++r) {
          float v = fmaxf((acc[t][r] - mu[r]) * rs[r] * g + be, 0.0f);
          outH[(m0 + hi * 8 + r) * Nout + n] = f2bf(v);
        }
      } else {
        float mx = 0.0f;                      // ReLU => column max >= 0
#pragma unroll
        for (int r = 0; r < 8; ++r) {
          float v = fmaxf((acc[t][r] - mu[r]) * rs[r] * g + be, 0.0f);
          mx = fmaxf(mx, v);
        }
        mx = fmaxf(mx, __shfl_xor(mx, 16, 32));  // merge the two M-halves
        if (hi == 0) atomicMax(&colmax[n], __float_as_uint(mx));
      }
    }
  }
}

__global__ __launch_bounds__(TPB) void localgraph_fused(
    const float* __restrict__ X,
    const float* __restrict__ W0, const float* __restrict__ b0,
    const float* __restrict__ g0, const float* __restrict__ be0,
    const float* __restrict__ W1, const float* __restrict__ b1,
    const float* __restrict__ g1, const float* __restrict__ be1,
    const float* __restrict__ W2, const float* __restrict__ b2,
    const float* __restrict__ g2, const float* __restrict__ be2,
    const float* __restrict__ W3, const float* __restrict__ b3,
    const float* __restrict__ g3, const float* __restrict__ be3,
    float* __restrict__ out) {
  // LDS layout (bytes):
  //   [0,32K)    H1 (256x64 bf16)    -- dead after L1, reused as low half of H3
  //   [0,64K)    H3 (256x128 bf16)
  //   [64K,96K)  H2 (256x64 bf16)
  //   [96K,128K) WL swizzled weights (<= 128x128 bf16)
  //   [128K+..]  m1/m2/i1 (64 each) + colmax[128]
  __shared__ __align__(16) unsigned char smem[135168];
  unsigned short* H1 = (unsigned short*)(smem);
  unsigned short* H3 = (unsigned short*)(smem);
  unsigned short* H2 = (unsigned short*)(smem + 65536);
  unsigned short* WL = (unsigned short*)(smem + 98304);
  float*    m1 = (float*)(smem + 131072);
  float*    m2 = (float*)(smem + 131328);
  int*      i1 = (int*)(smem + 131584);
  unsigned* colmax = (unsigned*)(smem + 131840);

  const int tid = threadIdx.x;
  const int blk = blockIdx.x;   // b*128 + m

  // ---- L0: X(256x8) @ W0(8x64) + b0 -> LN -> ReLU -> H1  (VALU; K=8 too small
  //      for WMMA). One thread per node row, fully in registers.
  {
    const float* xr = X + ((size_t)blk * 256 + tid) * 8;
    float x[8];
#pragma unroll
    for (int k = 0; k < 8; ++k) x[k] = xr[k];
    float h[64];
#pragma unroll
    for (int n = 0; n < 64; ++n) h[n] = b0[n];
#pragma unroll
    for (int k = 0; k < 8; ++k) {
#pragma unroll
      for (int n = 0; n < 64; ++n) h[n] += x[k] * W0[k * 64 + n];
    }
    float s = 0.0f, s2 = 0.0f;
#pragma unroll
    for (int n = 0; n < 64; ++n) { s += h[n]; s2 += h[n] * h[n]; }
    float mu  = s * (1.0f / 64.0f);
    float var = s2 * (1.0f / 64.0f) - mu * mu;
    float rs  = rsqrtf(var + 1e-5f);
#pragma unroll
    for (int n = 0; n < 64; ++n) {
      float v = fmaxf((h[n] - mu) * rs * g0[n] + be0[n], 0.0f);
      H1[tid * 64 + n] = f2bf(v);
    }
  }
  stage_weights<64, 64>(W1, WL);
  __syncthreads();

  // ---- L1: H1 @ W1 -> LN -> ReLU -> H2 ----
  mlp_stage<4, 2, 0>(H1, 64, WL, b1, g1, be1, H2, nullptr, nullptr, nullptr,
                     nullptr);
  __syncthreads();

  // ---- top-2 over nodes per feature (exclude-self-max), W2 staging, init ----
  if (tid < 64) {
    float v1 = -3.4e38f, v2 = -3.4e38f;
    int idx = 0;
    for (int i = 0; i < 256; ++i) {
      float v = bf2f(H2[i * 64 + tid]);
      if (v > v1)      { v2 = v1; v1 = v; idx = i; }
      else if (v > v2) { v2 = v; }
    }
    m1[tid] = v1; m2[tid] = v2; i1[tid] = idx;
  }
  if (tid < 128) colmax[tid] = 0u;
  stage_weights<128, 128>(W2, WL);
  __syncthreads();

  // ---- L2: concat([H2, exclmax(H2)]) @ W2 -> LN -> ReLU -> H3
  //      (concat half synthesized on the fly while loading A tiles) ----
  mlp_stage<8, 4, 1>(H2, 64, WL, b2, g2, be2, H3, m1, m2, i1, nullptr);
  __syncthreads();
  stage_weights<128, 128>(W3, WL);
  __syncthreads();

  // ---- L3: H3 @ W3 -> LN -> ReLU -> column max over the 256 nodes ----
  // Final concat half is algebraically identical to the column max (post-ReLU
  // values >= 0 and << 1e4 make the -1e4 branch dead and max_i excl_i == m1).
  mlp_stage<8, 4, 0>(H3, 128, WL, b3, g3, be3, nullptr, nullptr, nullptr,
                     nullptr, colmax);
  __syncthreads();

  if (tid < 128) {
    float v = __uint_as_float(colmax[tid]);
    float* o = out + (size_t)blk * 256;
    o[tid]       = v;   // first concat half
    o[tid + 128] = v;   // exclude-self-max half == column max (see analysis)
  }
}

extern "C" void kernel_launch(void* const* d_in, const int* in_sizes, int n_in,
                              void* d_out, int out_size, void* d_ws,
                              size_t ws_size, hipStream_t stream) {
  // setup_inputs() dict order: W0,b0,g0,be0, W1,b1,g1,be1, W2,..., W3,..., input_states
  const float* W0  = (const float*)d_in[0];
  const float* b0  = (const float*)d_in[1];
  const float* g0  = (const float*)d_in[2];
  const float* be0 = (const float*)d_in[3];
  const float* W1  = (const float*)d_in[4];
  const float* b1  = (const float*)d_in[5];
  const float* g1  = (const float*)d_in[6];
  const float* be1 = (const float*)d_in[7];
  const float* W2  = (const float*)d_in[8];
  const float* b2  = (const float*)d_in[9];
  const float* g2  = (const float*)d_in[10];
  const float* be2 = (const float*)d_in[11];
  const float* W3  = (const float*)d_in[12];
  const float* b3  = (const float*)d_in[13];
  const float* g3  = (const float*)d_in[14];
  const float* be3 = (const float*)d_in[15];
  const float* X   = (const float*)d_in[16];
  float* out = (float*)d_out;

  localgraph_fused<<<dim3(8 * 128), dim3(TPB), 0, stream>>>(
      X, W0, b0, g0, be0, W1, b1, g1, be1, W2, b2, g2, be2, W3, b3, g3, be3,
      out);
}
